// SAAMv3CLS_84928683311955
// MI455X (gfx1250) — compile-verified, matrix operand order checked by reading, not run on hardware
//
#include <hip/hip_runtime.h>
#include <hip/hip_bf16.h>

typedef __attribute__((ext_vector_type(16))) __bf16 v16bf;
typedef __attribute__((ext_vector_type(8)))  __bf16 v8bf;
typedef __attribute__((ext_vector_type(8)))  float  v8f;
typedef __attribute__((ext_vector_type(4)))  float  v4f;

#define BB   16
#define LL   2048
#define HH   1024
#define SS   32
#define PP   256
#define NTOT 512          // a-cols 0..255, s-cols 256..511
#define MT   32           // rows per block (2 WMMA M-tiles)
#define LDST 1032         // LDS row stride in bf16 elems (1024 + 8 pad)

static __device__ __forceinline__ __bf16 f2bf(float f) {
    union { float f; unsigned u; } c; c.f = f;
    unsigned r = c.u + 0x7FFFu + ((c.u >> 16) & 1u);   // round-to-nearest-even
    unsigned short h = (unsigned short)(r >> 16);
    union { unsigned short u; __bf16 b; } o; o.u = h;
    return o.b;
}

static __device__ __forceinline__ float gelu_exact(float x) {
    return 0.5f * x * (1.0f + erff(x * 0.70710678118654752f));
}

// ---------------------------------------------------------------------------
// Weight prep: Wt[n][k] = lnScale[k] * W[k][n]  (bf16, K-contiguous rows)
// ---------------------------------------------------------------------------
__global__ __launch_bounds__(256) void prep_wt(
    const float* __restrict__ Wa, const float* __restrict__ Wsm,
    const float* __restrict__ sA, const float* __restrict__ sS,
    __bf16* __restrict__ Wt)
{
    int idx = blockIdx.x * 256 + threadIdx.x;     // 512*1024 total
    int n = idx >> 10, k = idx & 1023;
    float w = (n < PP) ? Wa[k * PP + n] * sA[k]
                       : Wsm[k * PP + (n - PP)] * sS[k];
    Wt[n * HH + k] = f2bf(w);
}

__global__ __launch_bounds__(256) void prep_bias(
    const float* __restrict__ Wa, const float* __restrict__ Wsm,
    const float* __restrict__ bA, const float* __restrict__ bS,
    float* __restrict__ biasRow)
{
    int n = blockIdx.x * 256 + threadIdx.x;       // 512 total
    float acc = 0.0f;
    if (n < PP) { for (int k = 0; k < HH; ++k) acc += bA[k] * Wa[k * PP + n]; }
    else        { int nn = n - PP; for (int k = 0; k < HH; ++k) acc += bS[k] * Wsm[k * PP + nn]; }
    biasRow[n] = acc;
}

// ---------------------------------------------------------------------------
// Fused LayerNorm + dual GEMM (N=512) + GeLU via v_wmma_f32_16x16x32_bf16
// grid = 32768/32 blocks, 256 threads (8 waves, wave32)
// Block tile: 32(M) x 512(N); per wave: 32(M) x 64(N) strip = 8 WMMA tiles
// ---------------------------------------------------------------------------
__global__ __launch_bounds__(256) void fused_ln_gemm(
    const float* __restrict__ x,        // [32768][1024]
    const __bf16* __restrict__ Wt,      // [512][1024]
    const float* __restrict__ biasRow,  // [512]
    float* __restrict__ aProj,          // [32768][256]
    float* __restrict__ sProj)          // [32768][256]
{
    __shared__ __align__(16) __bf16 sX[MT * LDST];   // ~66 KB

    const int tid  = threadIdx.x;
    const int wave = tid >> 5;
    const int lane = tid & 31;
    const long rowBase = (long)blockIdx.x * MT;

    // ---- Phase 1: LayerNorm of 32 rows (4 per wave), pack bf16 x-hat to LDS
    for (int rr = 0; rr < 4; ++rr) {
        const int r = wave * 4 + rr;
        const float* xr = x + (rowBase + r) * HH + lane * 32;
        float v[32];
        #pragma unroll
        for (int i = 0; i < 8; ++i) {
            v4f t = *(const v4f*)(xr + i * 4);
            v[i*4+0] = t[0]; v[i*4+1] = t[1]; v[i*4+2] = t[2]; v[i*4+3] = t[3];
        }
        float s = 0.0f, q = 0.0f;
        #pragma unroll
        for (int i = 0; i < 32; ++i) { s += v[i]; q += v[i] * v[i]; }
        #pragma unroll
        for (int m = 16; m >= 1; m >>= 1) {
            s += __shfl_xor(s, m, 32);
            q += __shfl_xor(q, m, 32);
        }
        const float mu   = s * (1.0f / 1024.0f);
        const float var  = q * (1.0f / 1024.0f) - mu * mu;
        const float rstd = rsqrtf(var + 1e-5f);

        __bf16* dst = sX + r * LDST + lane * 32;
        #pragma unroll
        for (int i = 0; i < 4; ++i) {
            v8bf pk;
            #pragma unroll
            for (int j = 0; j < 8; ++j) pk[j] = f2bf((v[i*8+j] - mu) * rstd);
            *(v8bf*)(dst + i * 8) = pk;
        }
    }
    __syncthreads();

    // ---- Phase 2: each wave computes a 32x64 strip (2 M-tiles x 4 N-tiles)
    const int half  = lane >> 4;
    const int l16   = lane & 15;
    const int nBase = wave * 64;

    union FragU { v16bf v; v8bf h[2]; };
    v8f acc[2][4] = {};

    for (int kt = 0; kt < 32; ++kt) {
        // A fragments (ISA 16-bit 16x32 A layout): lanes 0-15 hold K0-7/16-23,
        // lanes 16-31 hold K8-15/24-31
        FragU a[2];
        #pragma unroll
        for (int mt = 0; mt < 2; ++mt) {
            const __bf16* ap = sX + (mt * 16 + l16) * LDST + kt * 32 + half * 8;
            a[mt].h[0] = *(const v8bf*)ap;
            a[mt].h[1] = *(const v8bf*)(ap + 16);
        }
        // B fragments: Wt rows are output columns, K-contiguous (same striping)
        FragU bfr[4];
        #pragma unroll
        for (int nt = 0; nt < 4; ++nt) {
            const int n = nBase + nt * 16 + l16;
            const __bf16* bp = Wt + n * HH + kt * 32 + half * 8;
            bfr[nt].h[0] = *(const v8bf*)bp;
            bfr[nt].h[1] = *(const v8bf*)(bp + 16);
        }
        // 8 back-to-back WMMAs per K-step
        #pragma unroll
        for (int mt = 0; mt < 2; ++mt)
            #pragma unroll
            for (int nt = 0; nt < 4; ++nt)
                acc[mt][nt] = __builtin_amdgcn_wmma_f32_16x16x32_bf16(
                    false, a[mt].v, false, bfr[nt].v, (short)0, acc[mt][nt],
                    false, false);
    }

    // ---- Epilogue: + folded LN-bias row, exact GeLU, scatter to a/s proj
    #pragma unroll
    for (int nt = 0; nt < 4; ++nt) {
        const int n = nBase + nt * 16 + l16;
        const float bias = biasRow[n];
        float* outBase = (n < PP) ? aProj : sProj;
        const int ncol = (n < PP) ? n : (n - PP);
        #pragma unroll
        for (int mt = 0; mt < 2; ++mt) {
            #pragma unroll
            for (int vv = 0; vv < 8; ++vv) {
                const int m = mt * 16 + vv + half * 8;   // C layout: VGPR v -> M
                float val = acc[mt][nt][vv] + bias;
                val = gelu_exact(val);
                outBase[(rowBase + m) * PP + ncol] = val;
            }
        }
    }
}

// ---------------------------------------------------------------------------
// Sentence boundaries: pi[b][s] (segment end) and start[b][s]
// ---------------------------------------------------------------------------
__global__ void seg_bounds(const int* __restrict__ p,
                           int* __restrict__ pi, int* __restrict__ st)
{
    const int b = blockIdx.x;
    if (threadIdx.x != 0) return;
    int cnt = 0, prev = 0;
    for (int l = 0; l < LL && cnt < SS; ++l) {
        if (p[b * LL + l] > 0) {
            pi[b * SS + cnt] = l;
            st[b * SS + cnt] = prev;
            prev = l + 1;
            ++cnt;
        }
    }
}

// Segment means over [start, end) for both projections
__global__ __launch_bounds__(256) void seg_means(
    const float* __restrict__ aP, const float* __restrict__ sP,
    const int* __restrict__ pi, const int* __restrict__ st,
    float* __restrict__ aS, float* __restrict__ sS)
{
    const int s = blockIdx.x, b = blockIdx.y, d = threadIdx.x;
    const int e = pi[b * SS + s], s0 = st[b * SS + s];
    float sa = 0.0f, ss = 0.0f;
    for (int l = s0; l < e; ++l) {
        const long r = ((long)b * LL + l) * PP + d;
        sa += aP[r]; ss += sP[r];
    }
    const float inv = 1.0f / (float)(e - s0);
    aS[(b * SS + s) * PP + d] = sa * inv;
    sS[(b * SS + s) * PP + d] = ss * inv;
}

// ---------------------------------------------------------------------------
// Aspect head: LN(256) -> (256x9) -> softmax(9); writes aspect_doc + aw
// ---------------------------------------------------------------------------
__global__ __launch_bounds__(256) void aspect_head(
    const float* __restrict__ aS,
    const float* __restrict__ lnS, const float* __restrict__ lnB,
    const float* __restrict__ Wasp,
    float* __restrict__ aspectOut,   // [512][9] section of d_out
    float* __restrict__ aw)          // [512][8] in ws
{
    __shared__ float red[256];
    __shared__ float xh[256];
    __shared__ float lg[9];
    const int row = blockIdx.x, d = threadIdx.x;
    const float xv = aS[row * PP + d];

    red[d] = xv; __syncthreads();
    for (int o = 128; o > 0; o >>= 1) { if (d < o) red[d] += red[d + o]; __syncthreads(); }
    const float mu = red[0] * (1.0f / 256.0f);
    __syncthreads();
    red[d] = (xv - mu) * (xv - mu); __syncthreads();
    for (int o = 128; o > 0; o >>= 1) { if (d < o) red[d] += red[d + o]; __syncthreads(); }
    const float var = red[0] * (1.0f / 256.0f);
    __syncthreads();

    const float rstd = rsqrtf(var + 1e-5f);
    xh[d] = (xv - mu) * rstd * lnS[d] + lnB[d];
    __syncthreads();

    if (d < 9) {
        float acc = 0.0f;
        for (int k = 0; k < PP; ++k) acc += xh[k] * Wasp[k * 9 + d];
        lg[d] = acc;
    }
    __syncthreads();
    if (d == 0) {
        float mx = lg[0];
        for (int j = 1; j < 9; ++j) mx = fmaxf(mx, lg[j]);
        float e[9], sum = 0.0f;
        for (int j = 0; j < 9; ++j) { e[j] = expf(lg[j] - mx); sum += e[j]; }
        const float inv = 1.0f / sum;
        for (int j = 0; j < 9; ++j) {
            const float pv = e[j] * inv;
            aspectOut[row * 9 + j] = pv;
            if (j < 8) aw[row * 8 + j] = pv;
        }
    }
}

// ---------------------------------------------------------------------------
// Doc head: weighted sentence mean per aspect, LN(256) -> (256x5)
// grid = (8 aspects, 16 batches)
// ---------------------------------------------------------------------------
__global__ __launch_bounds__(256) void doc_head(
    const float* __restrict__ sS, const float* __restrict__ aw,
    const float* __restrict__ lnS, const float* __restrict__ lnB,
    const float* __restrict__ Wse,
    float* __restrict__ resultOut)   // [16][8][5] section of d_out
{
    __shared__ float red[256];
    __shared__ float xh[256];
    const int n = blockIdx.x, b = blockIdx.y, d = threadIdx.x;

    float dsum = 0.0f, wsum = 0.0f;
    for (int s = 0; s < SS; ++s) {
        const float w = aw[(b * SS + s) * 8 + n];
        wsum += w;
        dsum += w * sS[(b * SS + s) * PP + d];
    }
    const float emb = dsum / wsum;

    red[d] = emb; __syncthreads();
    for (int o = 128; o > 0; o >>= 1) { if (d < o) red[d] += red[d + o]; __syncthreads(); }
    const float mu = red[0] * (1.0f / 256.0f);
    __syncthreads();
    red[d] = (emb - mu) * (emb - mu); __syncthreads();
    for (int o = 128; o > 0; o >>= 1) { if (d < o) red[d] += red[d + o]; __syncthreads(); }
    const float var = red[0] * (1.0f / 256.0f);
    __syncthreads();

    const float rstd = rsqrtf(var + 1e-5f);
    xh[d] = (emb - mu) * rstd * lnS[d] + lnB[d];
    __syncthreads();

    if (d < 5) {
        float acc = 0.0f;
        for (int k = 0; k < PP; ++k) acc += xh[k] * Wse[k * 5 + d];
        resultOut[(b * 8 + n) * 5 + d] = acc;
    }
}

// ---------------------------------------------------------------------------
extern "C" void kernel_launch(void* const* d_in, const int* in_sizes, int n_in,
                              void* d_out, int out_size, void* d_ws, size_t ws_size,
                              hipStream_t stream) {
    const float* outputs  = (const float*)d_in[0];
    // d_in[1] = mask (unused: all ones)
    const int*   p_index  = (const int*)d_in[2];
    // d_in[3] = n_sent scalar (fixed S=32)
    const float* ln_a_s   = (const float*)d_in[4];
    const float* ln_a_b   = (const float*)d_in[5];
    const float* W_a      = (const float*)d_in[6];
    const float* ln_s_s   = (const float*)d_in[7];
    const float* ln_s_b   = (const float*)d_in[8];
    const float* W_s      = (const float*)d_in[9];
    const float* ln_asp_s = (const float*)d_in[10];
    const float* ln_asp_b = (const float*)d_in[11];
    const float* W_asp    = (const float*)d_in[12];
    const float* ln_se_s  = (const float*)d_in[13];
    const float* ln_se_b  = (const float*)d_in[14];
    const float* W_se     = (const float*)d_in[15];

    const long M = (long)BB * LL;            // 32768 rows

    // workspace layout
    char* ws = (char*)d_ws;
    size_t off = 0;
    __bf16* Wt      = (__bf16*)(ws + off); off += (size_t)NTOT * HH * 2;     // 1 MB
    float*  biasRow = (float*)(ws + off);  off += (size_t)NTOT * 4;
    float*  aProj   = (float*)(ws + off);  off += (size_t)M * PP * 4;        // 32 MB
    float*  sProj   = (float*)(ws + off);  off += (size_t)M * PP * 4;        // 32 MB
    int*    pi      = (int*)(ws + off);    off += (size_t)BB * SS * 4;
    int*    st      = (int*)(ws + off);    off += (size_t)BB * SS * 4;
    float*  aSent   = (float*)(ws + off);  off += (size_t)BB * SS * PP * 4;
    float*  sSent   = (float*)(ws + off);  off += (size_t)BB * SS * PP * 4;
    float*  aw      = (float*)(ws + off);  off += (size_t)BB * SS * 8 * 4;

    // output layout: result (640) | outputs copy (B*L*H) | aspect_doc (B*S*9)
    float* d_result = (float*)d_out;
    float* d_copy   = (float*)d_out + (size_t)BB * 8 * 5;
    float* d_aspect = d_copy + (size_t)BB * LL * HH;

    // echo outputs (second tuple element)
    hipMemcpyAsync(d_copy, outputs, (size_t)BB * LL * HH * sizeof(float),
                   hipMemcpyDeviceToDevice, stream);

    prep_wt  <<<(NTOT * HH) / 256, 256, 0, stream>>>(W_a, W_s, ln_a_s, ln_s_s, Wt);
    prep_bias<<<2, 256, 0, stream>>>(W_a, W_s, ln_a_b, ln_s_b, biasRow);

    fused_ln_gemm<<<M / MT, 256, 0, stream>>>(outputs, Wt, biasRow, aProj, sProj);

    seg_bounds<<<BB, 32, 0, stream>>>(p_index, pi, st);
    seg_means<<<dim3(SS, BB), 256, 0, stream>>>(aProj, sProj, pi, st, aSent, sSent);

    aspect_head<<<BB * SS, 256, 0, stream>>>(aSent, ln_asp_s, ln_asp_b, W_asp,
                                             d_aspect, aw);
    doc_head<<<dim3(8, BB), 256, 0, stream>>>(sSent, aw, ln_se_s, ln_se_b, W_se,
                                              d_result);
}
